// BahdanauAttention_76347338653800
// MI455X (gfx1250) — compile-verified, matrix operand order checked by reading, not run on hardware
//
#include <hip/hip_runtime.h>
#include <hip/hip_bf16.h>
#include <math.h>

#define NB 32
#define NS 4096
#define NH 512
#define NU 512

typedef __attribute__((ext_vector_type(16))) __bf16 v16bf;
typedef __attribute__((ext_vector_type(8)))  float  v8f;
typedef __attribute__((ext_vector_type(4)))  unsigned int uint32x4;
typedef __attribute__((ext_vector_type(4)))  int  int32x4;
typedef __attribute__((ext_vector_type(8)))  int  int32x8;

__device__ __forceinline__ unsigned short f2bf(float f) {
    unsigned int u = __float_as_uint(f);
    unsigned int r = u + 0x7FFFu + ((u >> 16) & 1u);   // RNE
    return (unsigned short)(r >> 16);
}

// k_local within a 32-wide K step for the 16-bit A/B WMMA lane layout
__device__ __forceinline__ int klocal(int hi, int j) {
    int v = j >> 1, half = j & 1;
    return (v < 4) ? (hi * 8 + 2 * v + half)
                   : (16 + hi * 8 + 2 * (v - 4) + half);
}

// ---------------------------------------------------------------------------
// TDM: DMA a [rows x 512] f32 tile (row stride 512 elems) from global into LDS.
// D# layout per CDNA5 ISA ch.8 (async_tensor). 2D tensor -> groups 2/3 zero.
// ---------------------------------------------------------------------------
#if __has_builtin(__builtin_amdgcn_tensor_load_to_lds)
#define HAVE_TDM 1
__device__ __forceinline__ void tdm_load_tile_f32(const float* gptr,
                                                  unsigned lds_byte_off)
{
    unsigned long long ga = (unsigned long long)(uintptr_t)gptr;

    uint32x4 g0;
    g0[0] = 1u;                                   // count=1, is_restore=0
    g0[1] = lds_byte_off;                         // lds_addr
    g0[2] = (unsigned)(ga & 0xFFFFFFFFu);         // global_addr[31:0]
    g0[3] = (unsigned)((ga >> 32) & 0x01FFFFFFu)  // global_addr[56:32]
            | (2u << 30);                         // type = 2 ("image")

    int32x8 g1;
    g1[0] = (int)(2u << 16);                      // data_size=4B; mask=0; no pad
    g1[1] = (int)(512u << 16);                    // tensor_dim0 = 512 (lo16)
    g1[2] = (int)(4096u << 16);                   // dim0 hi=0 | tensor_dim1=4096
    g1[3] = (int)(512u << 16);                    // dim1 hi=0 | tile_dim0=512
    g1[4] = 16;                                   // tile_dim1=16, tile_dim2=0
    g1[5] = 512;                                  // tensor_dim0_stride = 512
    g1[6] = 0;                                    // stride0 hi | stride1 lo
    g1[7] = 0;

    int32x4 g2 = {}, g3 = {};
#if __clang_major__ >= 23
    int32x8 g4 = {};
    __builtin_amdgcn_tensor_load_to_lds(g0, g1, g2, g3, g4, 0);
#else
    __builtin_amdgcn_tensor_load_to_lds(g0, g1, g2, g3, 0);
#endif
}
#else
#define HAVE_TDM 0
#endif

// ---------------------------------------------------------------------------
// Phase 1: qb[b,u] = query[b,:]@W1[:,u] + b1[u] + b2[u]
// ---------------------------------------------------------------------------
__global__ __launch_bounds__(256) void qproj_kernel(
    const float* __restrict__ query, const float* __restrict__ W1,
    const float* __restrict__ b1, const float* __restrict__ b2,
    float* __restrict__ qb)
{
    const int b = blockIdx.x, t = threadIdx.x;
    __shared__ float qs[NH];
    for (int i = t; i < NH; i += 256) qs[i] = query[b * NH + i];
    __syncthreads();
    for (int u = t; u < NU; u += 256) {
        float acc = b1[u] + b2[u];
        for (int h = 0; h < NH; ++h)
            acc = fmaf(qs[h], W1[h * NU + u], acc);
        qb[b * NU + u] = acc;
    }
}

// ---------------------------------------------------------------------------
// Phase 2a: pre-swizzle W2 (f32, [H,U]) into bf16 B-matrix lane layout:
// linear idx = ((ut*16 + kt)*32 + lane)*16 + j
// ---------------------------------------------------------------------------
__global__ __launch_bounds__(256) void pack_w2_kernel(
    const float* __restrict__ W2, unsigned short* __restrict__ w2p)
{
    int idx  = blockIdx.x * 256 + threadIdx.x;        // 262144 total
    int ut   = idx >> 13;
    int rem1 = idx & 8191;
    int kt   = rem1 >> 9;
    int rem  = rem1 & 511;
    int l = rem >> 4, j = rem & 15;
    int hi = l >> 4, n = l & 15;
    int k = kt * 32 + klocal(hi, j);
    int u = ut * 16 + n;
    w2p[idx] = f2bf(W2[k * NU + u]);
}

// ---------------------------------------------------------------------------
// Phase 2b: main kernel. Per block: one batch b, one 16-row s-tile.
//   vproj = bf16_wmma(values_tile, W2), score[s] = sum_u va[u]*tanh(qb+vproj)
// ---------------------------------------------------------------------------
__global__ __launch_bounds__(256) void vproj_scores_kernel(
    const float* __restrict__ values, const unsigned short* __restrict__ w2p,
    const float* __restrict__ qb, const float* __restrict__ va,
    float* __restrict__ scores)
{
    __shared__ float ldsF[16 * NH];                                  // 32 KB
    __shared__ __align__(32) unsigned short ldsA[16 * 32 * 16];      // 16 KB
    __shared__ float ldsScore[16];

    const int st = blockIdx.x;            // s tile (256 per batch)
    const int b  = blockIdx.y;
    const int t  = threadIdx.x;
    const int lane = t & 31, wave = t >> 5;

    const float* tile = values + ((size_t)b * NS + (size_t)st * 16) * NH;

#if HAVE_TDM
    // DMA the 16x512 f32 tile into LDS via the Tensor Data Mover.
    if (t == 0) {
        tdm_load_tile_f32(tile, (unsigned)(uintptr_t)&ldsF[0]);
        __builtin_amdgcn_s_wait_tensorcnt(0);
    }
    if (t < 16) ldsScore[t] = 0.f;
    __syncthreads();
#else
    const float4* src  = reinterpret_cast<const float4*>(tile);
    float4* dstF = reinterpret_cast<float4*>(ldsF);
    for (int i = t; i < 16 * NH / 4; i += 256) dstF[i] = src[i];
    if (t < 16) ldsScore[t] = 0.f;
    __syncthreads();
#endif

    // swizzle+convert into the A-matrix (16x32 bf16) lane layout per K-step
    for (int i = t; i < 16 * 32 * 16; i += 256) {
        int kt  = i >> 9;
        int rem = i & 511;
        int l = rem >> 4, j = rem & 15;
        int hi = l >> 4, m = l & 15;
        int k = kt * 32 + klocal(hi, j);
        ldsA[i] = f2bf(ldsF[m * NH + k]);
    }
    __syncthreads();

    float part[8];
#pragma unroll
    for (int r = 0; r < 8; ++r) part[r] = 0.f;

    const int n  = lane & 15;
    const int hi = lane >> 4;

    for (int uti = 0; uti < 4; ++uti) {              // 8 waves x 4 = 32 u-tiles
        const int ut = wave * 4 + uti;
        v8f acc = {};
        for (int kt = 0; kt < 16; ++kt) {            // K = 512 in steps of 32
            v16bf a  = *reinterpret_cast<const v16bf*>(&ldsA[(kt * 32 + lane) * 16]);
            v16bf bm = *reinterpret_cast<const v16bf*>(
                &w2p[(((size_t)ut * 16 + kt) * 32 + lane) * 16]);
            acc = __builtin_amdgcn_wmma_f32_16x16x32_bf16(
                false, a, false, bm, (short)0, acc, false, false);
        }
        const int u   = ut * 16 + n;
        const float qbu = qb[b * NU + u];
        const float vau = va[u];
#pragma unroll
        for (int r = 0; r < 8; ++r)
            part[r] = fmaf(vau, tanhf(qbu + acc[r]), part[r]);
    }

    // reduce over the 16 lanes that share each output row m = r + 8*hi
#pragma unroll
    for (int r = 0; r < 8; ++r) {
        float p = part[r];
        for (int off = 1; off < 16; off <<= 1) p += __shfl_xor(p, off, 32);
        if (n == 0) atomicAdd(&ldsScore[r + 8 * hi], p);   // ds_add_f32
    }
    __syncthreads();
    if (t < 16)
        scores[(size_t)b * NS + st * 16 + t] = ldsScore[t];
}

// ---------------------------------------------------------------------------
// Phase 3a: per-batch softmax stats (max, sum of exp)
// ---------------------------------------------------------------------------
__global__ __launch_bounds__(256) void softmax_stats_kernel(
    const float* __restrict__ scores, float* __restrict__ stats)
{
    const int b = blockIdx.x, t = threadIdx.x;
    __shared__ float red[256];
    float m = -INFINITY;
    for (int s = t; s < NS; s += 256) m = fmaxf(m, scores[(size_t)b * NS + s]);
    red[t] = m; __syncthreads();
    for (int o = 128; o > 0; o >>= 1) {
        if (t < o) red[t] = fmaxf(red[t], red[t + o]);
        __syncthreads();
    }
    m = red[0]; __syncthreads();
    float z = 0.f;
    for (int s = t; s < NS; s += 256) z += __expf(scores[(size_t)b * NS + s] - m);
    red[t] = z; __syncthreads();
    for (int o = 128; o > 0; o >>= 1) {
        if (t < o) red[t] += red[t + o];
        __syncthreads();
    }
    if (t == 0) { stats[2 * b] = m; stats[2 * b + 1] = red[0]; }
}

__global__ __launch_bounds__(256) void zero_ctx_kernel(float* __restrict__ out_ctx)
{
    int i = blockIdx.x * 256 + threadIdx.x;
    if (i < NB * NH) out_ctx[i] = 0.f;
}

// ---------------------------------------------------------------------------
// Phase 3b: attention weights + context accumulation (second pass over values)
// ---------------------------------------------------------------------------
__global__ __launch_bounds__(256) void attn_context_kernel(
    const float* __restrict__ values, const float* __restrict__ scores,
    const float* __restrict__ stats, float* __restrict__ out_attn,
    float* __restrict__ out_ctx)
{
    const int b = blockIdx.x, sc = blockIdx.y, t = threadIdx.x;
    __shared__ float w[512];
    const float m  = stats[2 * b];
    const float rz = 1.f / stats[2 * b + 1];
    const int  s0  = sc * 512;

    for (int i = t; i < 512; i += 256) {
        float wi = __expf(scores[(size_t)b * NS + s0 + i] - m) * rz;
        w[i] = wi;
        out_attn[(size_t)b * NS + s0 + i] = wi;
    }
    __syncthreads();

    float acc0 = 0.f, acc1 = 0.f;
    const float* vb = values + ((size_t)b * NS + s0) * NH;
    for (int s = 0; s < 512; ++s) {
        float ws = w[s];
        acc0 = fmaf(ws, vb[(size_t)s * NH + t],       acc0);
        acc1 = fmaf(ws, vb[(size_t)s * NH + t + 256], acc1);
    }
    atomicAdd(&out_ctx[b * NH + t],       acc0);
    atomicAdd(&out_ctx[b * NH + t + 256], acc1);
}

// ---------------------------------------------------------------------------
extern "C" void kernel_launch(void* const* d_in, const int* in_sizes, int n_in,
                              void* d_out, int out_size, void* d_ws, size_t ws_size,
                              hipStream_t stream)
{
    const float* query  = (const float*)d_in[0];
    const float* values = (const float*)d_in[1];
    const float* W1     = (const float*)d_in[2];
    const float* b1     = (const float*)d_in[3];
    const float* W2     = (const float*)d_in[4];
    const float* b2     = (const float*)d_in[5];
    const float* va     = (const float*)d_in[6];
    // d_in[7] = bva: softmax-invariant (raw scores never output), dropped.

    float* out_attn = (float*)d_out;               // [B, S, 1]
    float* out_ctx  = (float*)d_out + NB * NS;     // [B, H]

    char* ws = (char*)d_ws;
    float*          qb     = (float*)ws;                                   // 64 KB
    unsigned short* w2p    = (unsigned short*)(ws + (64 << 10));           // 512 KB
    float*          scores = (float*)(ws + (64 << 10) + (512 << 10));      // 512 KB
    float*          stats  = (float*)(ws + (64 << 10) + (1024 << 10));     // 256 B

    qproj_kernel<<<NB, 256, 0, stream>>>(query, W1, b1, b2, qb);
    pack_w2_kernel<<<(NH * NU) / 256, 256, 0, stream>>>(W2, w2p);

    dim3 g2(NS / 16, NB);
    vproj_scores_kernel<<<g2, 256, 0, stream>>>(values, w2p, qb, va, scores);

    softmax_stats_kernel<<<NB, 256, 0, stream>>>(scores, stats);
    zero_ctx_kernel<<<(NB * NH + 255) / 256, 256, 0, stream>>>(out_ctx);

    dim3 g3(NB, NS / 512);
    attn_context_kernel<<<g3, 256, 0, stream>>>(values, scores, stats,
                                                out_attn, out_ctx);
}
